// MultiHeadAttention_27419071218134
// MI455X (gfx1250) — matrix operand order repaired and checked by reading, not compiled, not run on hardware
//
#include <hip/hip_runtime.h>

typedef __attribute__((ext_vector_type(16))) __bf16 bf16x16;
typedef __attribute__((ext_vector_type(8)))  __bf16 bf16x8;
typedef __attribute__((ext_vector_type(8)))  float  f32x8;

#define B_  32
#define N_  577
#define C_  768
#define H_  12
#define DK  64
#define M_  (B_*N_)        /* 18464 rows of x, divisible by 16 */
#define MT  (M_/16)        /* 1154 row tiles */
#define NQ  592            /* padded query rows per head (37*16) */
#define NK  608            /* padded key rows per head (19*32) */
#define KBLK (NK/32)       /* 19 key blocks */
#define QT  ((N_+15)/16)   /* 37 query tiles */

__device__ __forceinline__ f32x8 wmma_bf16(bf16x16 a, bf16x16 b, f32x8 c) {
  return __builtin_amdgcn_wmma_f32_16x16x32_bf16(false, a, false, b, (short)0, c, false, false);
}
__device__ __forceinline__ bf16x16 cat16(bf16x8 lo, bf16x8 hi) {
  return __builtin_shufflevector(lo, hi, 0,1,2,3,4,5,6,7,8,9,10,11,12,13,14,15);
}

// ---------------- fp32 -> bf16 conversion ----------------
__global__ void cvt_bf16(const float* __restrict__ s, __bf16* __restrict__ d, int n) {
  int i = (blockIdx.x * 256 + threadIdx.x) * 4;
  if (i + 3 < n) {
    float4 v = *(const float4*)(s + i);
    d[i+0] = (__bf16)v.x; d[i+1] = (__bf16)v.y;
    d[i+2] = (__bf16)v.z; d[i+3] = (__bf16)v.w;
  }
}

// ---------------- QKV projection: one wave -> 16x64 tile ----------------
// A = x_bf [M_, C_] row-major. B = W^T (W stored [out,in] row-major, so
// B[k][col] = W[col][k] is K-contiguous per lane -> contiguous loads).
__global__ __launch_bounds__(32) void qkv_gemm(
    const __bf16* __restrict__ xbf,
    const __bf16* __restrict__ Wqb, const __bf16* __restrict__ Wkb, const __bf16* __restrict__ Wvb,
    const float* __restrict__ bq, const float* __restrict__ bk, const float* __restrict__ bv,
    __bf16* __restrict__ Qb, __bf16* __restrict__ Kb, __bf16* __restrict__ Vt)
{
  const int mt = blockIdx.x;     // row tile
  const int hh = blockIdx.y;     // head == 64-col tile (768/64 == 12 == H)
  const int mat = blockIdx.z;    // 0=Q 1=K 2=V
  const int l = threadIdx.x, lid = l & 15, hf = l >> 4;

  const __bf16* W    = (mat == 0) ? Wqb : ((mat == 1) ? Wkb : Wvb);
  const float*  bias = (mat == 0) ? bq  : ((mat == 1) ? bk  : bv);

  // A fragment source: lane (l&15) = row, chunks at k+hf*16 and k+hf*16+8
  const __bf16* arow = xbf + (size_t)(mt*16 + lid) * C_ + hf*16;
  // B fragment sources: lane col = lid within each 16-col subtile, 16 K per lane
  const __bf16* bp0 = W + (size_t)(hh*64 +  0 + lid) * C_ + hf*16;
  const __bf16* bp1 = W + (size_t)(hh*64 + 16 + lid) * C_ + hf*16;
  const __bf16* bp2 = W + (size_t)(hh*64 + 32 + lid) * C_ + hf*16;
  const __bf16* bp3 = W + (size_t)(hh*64 + 48 + lid) * C_ + hf*16;

  f32x8 c0 = {0,0,0,0,0,0,0,0}, c1 = {0,0,0,0,0,0,0,0};
  f32x8 c2 = {0,0,0,0,0,0,0,0}, c3 = {0,0,0,0,0,0,0,0};

  for (int k = 0; k < C_; k += 32) {
    bf16x16 a = cat16(*(const bf16x8*)(arow + k), *(const bf16x8*)(arow + k + 8));
    c0 = wmma_bf16(a, *(const bf16x16*)(bp0 + k), c0);
    c1 = wmma_bf16(a, *(const bf16x16*)(bp1 + k), c1);
    c2 = wmma_bf16(a, *(const bf16x16*)(bp2 + k), c2);
    c3 = wmma_bf16(a, *(const bf16x16*)(bp3 + k), c3);
  }

  const float bb0 = bias[hh*64 +  0 + lid];
  const float bb1 = bias[hh*64 + 16 + lid];
  const float bb2 = bias[hh*64 + 32 + lid];
  const float bb3 = bias[hh*64 + 48 + lid];

  #pragma unroll
  for (int i = 0; i < 8; ++i) {
    const int rg = mt*16 + i + hf*8;     // global x row
    const int n  = rg % N_;
    const int bb = rg / N_;
    const int bh = bb * H_ + hh;
    if (mat == 0) {
      __bf16* q = Qb + ((size_t)bh * NQ + n) * DK;
      q[     lid] = (__bf16)(c0[i] + bb0);
      q[16 + lid] = (__bf16)(c1[i] + bb1);
      q[32 + lid] = (__bf16)(c2[i] + bb2);
      q[48 + lid] = (__bf16)(c3[i] + bb3);
    } else if (mat == 1) {
      __bf16* kk = Kb + ((size_t)bh * NK + n) * DK;
      kk[     lid] = (__bf16)(c0[i] + bb0);
      kk[16 + lid] = (__bf16)(c1[i] + bb1);
      kk[32 + lid] = (__bf16)(c2[i] + bb2);
      kk[48 + lid] = (__bf16)(c3[i] + bb3);
    } else {
      __bf16* v = Vt + (size_t)bh * DK * NK + n;   // transposed [64][NK]
      v[(size_t)(     lid) * NK] = (__bf16)(c0[i] + bb0);
      v[(size_t)(16 + lid) * NK] = (__bf16)(c1[i] + bb1);
      v[(size_t)(32 + lid) * NK] = (__bf16)(c2[i] + bb2);
      v[(size_t)(48 + lid) * NK] = (__bf16)(c3[i] + bb3);
    }
  }
}

// ---------------- flash attention: one wave -> 16 query rows x dk=64 ----------------
__global__ __launch_bounds__(32) void attn_kernel(
    const __bf16* __restrict__ Qb, const __bf16* __restrict__ Kb,
    const __bf16* __restrict__ Vt, __bf16* __restrict__ Ob)
{
  const int qt = blockIdx.x;   // 0..36
  const int h  = blockIdx.y;   // 0..11
  const int b  = blockIdx.z;   // 0..31
  const int l = threadIdx.x, lid = l & 15, hf = l >> 4;
  const int bh = b * H_ + h;

  // Q A-fragments: a0 = dk 0..31, a1 = dk 32..63
  const __bf16* qrow = Qb + ((size_t)bh * NQ + qt*16 + lid) * DK;
  const int ko = hf * 16;
  bf16x16 a0 = cat16(*(const bf16x8*)(qrow + ko),      *(const bf16x8*)(qrow + ko + 8));
  bf16x16 a1 = cat16(*(const bf16x8*)(qrow + 32 + ko), *(const bf16x8*)(qrow + 32 + ko + 8));

  f32x8 o0 = {0,0,0,0,0,0,0,0}, o1 = {0,0,0,0,0,0,0,0};
  f32x8 o2 = {0,0,0,0,0,0,0,0}, o3 = {0,0,0,0,0,0,0,0};
  float mi[8], li[8];
  #pragma unroll
  for (int i = 0; i < 8; ++i) { mi[i] = -1e30f; li[i] = 0.0f; }

  __shared__ __align__(32) __bf16 plds[16 * 32];

  for (int kb = 0; kb < KBLK; ++kb) {
    const int key0 = kb * 32;
    // K^T B-fragments: lane col = key, 16 consecutive dk per lane (K-contiguous)
    const __bf16* kc0 = Kb + ((size_t)bh * NK + key0 +      lid) * DK + hf*16;
    const __bf16* kc1 = Kb + ((size_t)bh * NK + key0 + 16 + lid) * DK + hf*16;
    f32x8 s0 = {0,0,0,0,0,0,0,0}, s1 = {0,0,0,0,0,0,0,0};
    s0 = wmma_bf16(a0, *(const bf16x16*)(kc0),      s0);
    s0 = wmma_bf16(a1, *(const bf16x16*)(kc0 + 32), s0);
    s1 = wmma_bf16(a0, *(const bf16x16*)(kc1),      s1);
    s1 = wmma_bf16(a1, *(const bf16x16*)(kc1 + 32), s1);

    const bool msk0 = (key0 +      lid) >= N_;
    const bool msk1 = (key0 + 16 + lid) >= N_;
    #pragma unroll
    for (int i = 0; i < 8; ++i) {
      s0[i] = msk0 ? -1e30f : s0[i] * 0.125f;   // 1/sqrt(64)
      s1[i] = msk1 ? -1e30f : s1[i] * 0.125f;
    }

    float p0[8], p1[8], alpha[8];
    #pragma unroll
    for (int i = 0; i < 8; ++i) {
      float v = fmaxf(s0[i], s1[i]);
      v = fmaxf(v, __shfl_xor(v, 1));
      v = fmaxf(v, __shfl_xor(v, 2));
      v = fmaxf(v, __shfl_xor(v, 4));
      v = fmaxf(v, __shfl_xor(v, 8));          // row max within 16-lane half
      const float mn = fmaxf(mi[i], v);
      alpha[i] = __expf(mi[i] - mn);
      mi[i] = mn;
      p0[i] = __expf(s0[i] - mn);
      p1[i] = __expf(s1[i] - mn);
      float r = p0[i] + p1[i];
      r += __shfl_xor(r, 1);
      r += __shfl_xor(r, 2);
      r += __shfl_xor(r, 4);
      r += __shfl_xor(r, 8);                   // row sum within 16-lane half
      li[i] = li[i] * alpha[i] + r;
    }
    #pragma unroll
    for (int i = 0; i < 8; ++i) {
      o0[i] *= alpha[i]; o1[i] *= alpha[i]; o2[i] *= alpha[i]; o3[i] *= alpha[i];
    }

    // transpose P (16x32, C-layout) to A-fragment layout through LDS
    __syncthreads();
    #pragma unroll
    for (int i = 0; i < 8; ++i) {
      const int row = i + hf * 8;
      plds[row * 32 +      lid] = (__bf16)p0[i];
      plds[row * 32 + 16 + lid] = (__bf16)p1[i];
    }
    __syncthreads();
    const __bf16* pr = plds + lid * 32 + hf * 16;
    bf16x16 pa = cat16(*(const bf16x8*)(pr), *(const bf16x8*)(pr + 8));

    // V B-fragments from transposed V: lane col = dk, 16 consecutive keys per lane
    const __bf16* vb = Vt + ((size_t)bh * DK + lid) * NK + key0 + hf*16;
    o0 = wmma_bf16(pa, *(const bf16x16*)(vb),              o0);
    o1 = wmma_bf16(pa, *(const bf16x16*)(vb + 16*(size_t)NK), o1);
    o2 = wmma_bf16(pa, *(const bf16x16*)(vb + 32*(size_t)NK), o2);
    o3 = wmma_bf16(pa, *(const bf16x16*)(vb + 48*(size_t)NK), o3);
  }

  // normalize + store to [b*N_+n][C_] with col = h*64+d
  #pragma unroll
  for (int i = 0; i < 8; ++i) {
    const int n = qt*16 + i + hf*8;
    if (n < N_) {
      const float inv = 1.0f / li[i];
      __bf16* o = Ob + ((size_t)b * N_ + n) * C_ + h * DK;
      o[     lid] = (__bf16)(o0[i] * inv);
      o[16 + lid] = (__bf16)(o1[i] * inv);
      o[32 + lid] = (__bf16)(o2[i] * inv);
      o[48 + lid] = (__bf16)(o3[i] * inv);
    }
  }
}

// ---------------- output projection -> fp32 d_out ----------------
__global__ __launch_bounds__(32) void out_proj(
    const __bf16* __restrict__ Ob, const __bf16* __restrict__ Wob,
    const float* __restrict__ bo, float* __restrict__ out)
{
  const int mt = blockIdx.x;   // row tile
  const int ct = blockIdx.y;   // 64-col tile
  const int l = threadIdx.x, lid = l & 15, hf = l >> 4;

  const __bf16* arow = Ob + (size_t)(mt*16 + lid) * C_ + hf*16;
  const __bf16* bp0 = Wob + (size_t)(ct*64 +  0 + lid) * C_ + hf*16;
  const __bf16* bp1 = Wob + (size_t)(ct*64 + 16 + lid) * C_ + hf*16;
  const __bf16* bp2 = Wob + (size_t)(ct*64 + 32 + lid) * C_ + hf*16;
  const __bf16* bp3 = Wob + (size_t)(ct*64 + 48 + lid) * C_ + hf*16;

  f32x8 c0 = {0,0,0,0,0,0,0,0}, c1 = {0,0,0,0,0,0,0,0};
  f32x8 c2 = {0,0,0,0,0,0,0,0}, c3 = {0,0,0,0,0,0,0,0};

  for (int k = 0; k < C_; k += 32) {
    bf16x16 a = cat16(*(const bf16x8*)(arow + k), *(const bf16x8*)(arow + k + 8));
    c0 = wmma_bf16(a, *(const bf16x16*)(bp0 + k), c0);
    c1 = wmma_bf16(a, *(const bf16x16*)(bp1 + k), c1);
    c2 = wmma_bf16(a, *(const bf16x16*)(bp2 + k), c2);
    c3 = wmma_bf16(a, *(const bf16x16*)(bp3 + k), c3);
  }

  const float bb0 = bo[ct*64 +  0 + lid];
  const float bb1 = bo[ct*64 + 16 + lid];
  const float bb2 = bo[ct*64 + 32 + lid];
  const float bb3 = bo[ct*64 + 48 + lid];

  #pragma unroll
  for (int i = 0; i < 8; ++i) {
    const size_t rg = (size_t)(mt*16 + i + hf*8);
    float* o = out + rg * C_ + ct * 64;
    o[     lid] = c0[i] + bb0;
    o[16 + lid] = c1[i] + bb1;
    o[32 + lid] = c2[i] + bb2;
    o[48 + lid] = c3[i] + bb3;
  }
}

extern "C" void kernel_launch(void* const* d_in, const int* in_sizes, int n_in,
                              void* d_out, int out_size, void* d_ws, size_t ws_size,
                              hipStream_t stream) {
  (void)in_sizes; (void)n_in; (void)out_size; (void)ws_size;
  const float* x  = (const float*)d_in[0];
  const float* Wq = (const float*)d_in[1];
  const float* bq = (const float*)d_in[2];
  const float* Wk = (const float*)d_in[3];
  const float* bk = (const float*)d_in[4];
  const float* Wv = (const float*)d_in[5];
  const float* bv = (const float*)d_in[6];
  const float* Wo = (const float*)d_in[7];
  const float* bo = (const float*)d_in[8];

  char* ws = (char*)d_ws;
  const size_t WSZ = (size_t)C_ * C_ * 2;           // bf16 weight matrix bytes
  __bf16* Wqb = (__bf16*)(ws + 0*WSZ);
  __bf16* Wkb = (__bf16*)(ws + 1*WSZ);
  __bf16* Wvb = (__bf16*)(ws + 2*WSZ);
  __bf16* Wob = (__bf16*)(ws + 3*WSZ);
  __bf16* xb  = (__bf16*)(ws + 4*WSZ);
  size_t off = 4*WSZ + (size_t)M_ * C_ * 2;
  __bf16* Qb = (__bf16*)(ws + off); off += (size_t)B_ * H_ * NQ * DK * 2;
  __bf16* Kb = (__bf16*)(ws + off); off += (size_t)B_ * H_ * NK * DK * 2;
  __bf16* Vt = (__bf16*)(ws + off); off += (size_t)B_ * H_ * NK * DK * 2;
  __bf16* Ob = (__bf16*)(ws + off);

  const int nw = C_ * C_;
  cvt_bf16<<<(nw/4 + 255)/256, 256, 0, stream>>>(Wq, Wqb, nw);
  cvt_bf16<<<(nw/4 + 255)/256, 256, 0, stream>>>(Wk, Wkb, nw);
  cvt_bf16<<<(nw/4 + 255)/256, 256, 0, stream>>>(Wv, Wvb, nw);
  cvt_bf16<<<(nw/4 + 255)/256, 256, 0, stream>>>(Wo, Wob, nw);
  const int nx = M_ * C_;
  cvt_bf16<<<(nx/4 + 255)/256, 256, 0, stream>>>(x, xb, nx);

  qkv_gemm<<<dim3(MT, H_, 3), 32, 0, stream>>>(xb, Wqb, Wkb, Wvb, bq, bk, bv, Qb, Kb, Vt);
  attn_kernel<<<dim3(QT, H_, B_), 32, 0, stream>>>(Qb, Kb, Vt, Ob);
  out_proj<<<dim3(MT, C_/64), 32, 0, stream>>>(Ob, Wob, bo, (float*)d_out);
}
